// Paulick2024_4148938408294
// MI455X (gfx1250) — matrix operand order repaired and checked by reading, not compiled, not run on hardware
//
#include <hip/hip_runtime.h>
#include <hip/hip_bf16.h>
#include <math.h>

// ---------------------------------------------------------------------------
// Paulick 2024 auditory model, CDNA5 (gfx1250) implementation.
// Stage 1 (DRNL gammatone FIR bank) is an implicit-GEMM on
// v_wmma_f32_16x16x32_f16 with filter tiles staged into LDS by the
// Tensor Data Mover (tensor_load_to_lds + s_wait_tensorcnt).
// ---------------------------------------------------------------------------

#define FS_F 44100.0f
#define N_CH 50
#define TLEN 88200
#define GTL 2048
#define TTILE 128
#define NCHP 64                      // padded filter rows per bank
#define TP (GTL + TTILE * 690)       // padded f16 signal length = 90368
#define T2LEN 22050
#define PI_F 3.14159265358979f
#define TWO_PI_F 6.28318530717959f

typedef __attribute__((ext_vector_type(16))) _Float16 v16h;
typedef __attribute__((ext_vector_type(8)))  _Float16 v8h;
typedef __attribute__((ext_vector_type(8)))  float    v8f;
typedef __attribute__((ext_vector_type(4)))  unsigned int u32x4;
typedef __attribute__((ext_vector_type(8)))  int          i32x8;
typedef __attribute__((ext_vector_type(4)))  int          i32x4;

#if defined(__gfx1250__) && __has_builtin(__builtin_amdgcn_tensor_load_to_lds)
#define HAVE_TDM 1
#else
#define HAVE_TDM 0
#endif

// ---- erbspace center frequency for channel c ------------------------------
__device__ __forceinline__ float fcenter(int c) {
    const float k = 4.37e-3f;
    float elo = 21.4f * log10f(k * 125.0f + 1.0f);
    float ehi = 21.4f * log10f(k * 8000.0f + 1.0f);
    float e = elo + (ehi - elo) * ((float)c / 49.0f);
    return (powf(10.0f, e / 21.4f) - 1.0f) / k;
}

#if HAVE_TDM
// low 32 bits of a flat shared-aperture address are the LDS byte offset
__device__ __forceinline__ unsigned lds_addr_of(const void* p) {
    return (unsigned)(unsigned long long)(uintptr_t)p;
}

// 2D strided tile (tile_w x tile_h elements, 2-byte elems) global -> LDS
__device__ __forceinline__ void tdm_load_2d_f16(unsigned lds_off,
                                                const void* gaddr,
                                                unsigned tile_w, unsigned tile_h,
                                                unsigned tensor_w, unsigned tensor_h,
                                                unsigned row_stride_elems) {
    unsigned long long ga = (unsigned long long)(uintptr_t)gaddr;
    u32x4 g0;
    g0[0] = 1u;                                            // count=1, user D#
    g0[1] = lds_off;                                       // lds_addr (bytes)
    g0[2] = (unsigned)(ga & 0xFFFFFFFFu);                  // global_addr[31:0]
    g0[3] = ((unsigned)(ga >> 32) & 0x01FFFFFFu)           // global_addr[56:32]
            | 0x80000000u;                                 // type = 2 (image)
    i32x8 g1;
    g1[0] = (int)(1u << 16);                               // data_size=1 (2B)
    g1[1] = (int)((tensor_w & 0xFFFFu) << 16);             // tensor_dim0[15:0]
    g1[2] = (int)(((tensor_w >> 16) & 0xFFFFu) |
                  ((tensor_h & 0xFFFFu) << 16));           // dim0 hi | dim1 lo
    g1[3] = (int)(((tensor_h >> 16) & 0xFFFFu) |
                  ((tile_w & 0xFFFFu) << 16));             // dim1 hi | tile_dim0
    g1[4] = (int)(tile_h & 0xFFFFu);                       // tile_dim1 (dim2=0)
    g1[5] = (int)row_stride_elems;                         // dim0_stride[31:0]
    g1[6] = 0;                                             // stride hi / dim1_stride
    g1[7] = 0;
    i32x4 z4 = {0, 0, 0, 0};
#if __clang_major__ >= 23
    i32x8 z8 = {0, 0, 0, 0, 0, 0, 0, 0};
    __builtin_amdgcn_tensor_load_to_lds(g0, g1, z4, z4, z8, 0);
#else
    __builtin_amdgcn_tensor_load_to_lds(g0, g1, z4, z4, 0);
#endif
}
#endif // HAVE_TDM

// ===========================================================================
// Kernel 1: per-channel DRNL parameters -> params[c*8 + j]
//   j: 0=fc  1=g_lin  2=a_nl  3=b_nl  4=fc_nl
// ===========================================================================
__global__ void param_gen(float* __restrict__ params) {
    int c = threadIdx.x;
    if (c >= N_CH) return;
    float fcv = fcenter(c);
    float lf = log10f(fcv);
    params[c * 8 + 0] = fcv;
    params[c * 8 + 1] = powf(10.0f, 4.20405f - 0.47909f * lf);
    params[c * 8 + 2] = powf(10.0f, 1.40298f + 0.81916f * lf);
    params[c * 8 + 3] = powf(10.0f, 1.61912f - 0.81867f * lf);
    params[c * 8 + 4] = powf(10.0f, -0.05252f + 1.01650f * lf);
    params[c * 8 + 5] = 0.0f;
    params[c * 8 + 6] = 0.0f;
    params[c * 8 + 7] = 0.0f;
}

// ===========================================================================
// Kernel 2: generate normalized gammatone FIRs in f16.
// h16 layout: [bank(2)][row(64)][tap(2048)], rows >= 50 zeroed.
// ===========================================================================
__global__ __launch_bounds__(256) void fir_gen(_Float16* __restrict__ h16) {
    const int c = blockIdx.x;       // 0..63
    const int bank = blockIdx.y;    // 0 = lin, 1 = nl
    const int tid = threadIdx.x;
    _Float16* dst = h16 + ((size_t)(bank * NCHP + c)) * GTL;

    if (c >= N_CH) {
        for (int l = tid; l < GTL; l += 256) dst[l] = (_Float16)0.0f;
        return;
    }

    float fcv = fcenter(c);
    float lf = log10f(fcv);
    float fcg = (bank == 0) ? powf(10.0f, -0.06762f + 1.01679f * lf)
                            : powf(10.0f, -0.05252f + 1.01650f * lf);
    float bw  = (bank == 0) ? powf(10.0f,  0.03728f + 0.78563f * lf)
                            : powf(10.0f, -0.03193f + 0.77426f * lf);

    // complex response at fc: H = sum g[l] * exp(-2j*pi*fc*l/fs)
    float sre = 0.0f, sim = 0.0f;
    for (int l = tid; l < GTL; l += 256) {
        float t = (float)l / FS_F;
        float cosv = cosf(TWO_PI_F * fcg * t);
        float g = t * t * t * expf(-TWO_PI_F * bw * t) * cosv;
        sre += g * cosv;
        sim -= g * sinf(TWO_PI_F * fcg * t);
    }
    __shared__ float red[512];
    red[tid] = sre;
    red[256 + tid] = sim;
    __syncthreads();
    for (int s = 128; s > 0; s >>= 1) {
        if (tid < s) {
            red[tid] += red[tid + s];
            red[256 + tid] += red[256 + tid + s];
        }
        __syncthreads();
    }
    __shared__ float norm;
    if (tid == 0)
        norm = rsqrtf(red[0] * red[0] + red[256] * red[256] + 1e-30f);
    __syncthreads();
    float nv = norm;
    for (int l = tid; l < GTL; l += 256) {
        float t = (float)l / FS_F;
        float g = t * t * t * expf(-TWO_PI_F * bw * t) * cosf(TWO_PI_F * fcg * t);
        dst[l] = (_Float16)(g * nv);
    }
}

// ===========================================================================
// Kernel 3: f16-convert x with a GTL-sample zero prologue (causal conv)
// xh layout: [b][TP]
// ===========================================================================
__global__ __launch_bounds__(256) void xpad(const float* __restrict__ x,
                                            _Float16* __restrict__ xh) {
    int idx = blockIdx.x * 256 + threadIdx.x;
    if (idx >= 4 * TP) return;
    int b = idx / TP;
    int j = idx - b * TP;
    int t = j - GTL;
    float v = (t >= 0 && t < TLEN) ? x[(size_t)b * TLEN + t] : 0.0f;
    xh[idx] = (_Float16)v;
}

// ===========================================================================
// Kernel 4: DRNL FIR bank as WMMA implicit-GEMM.
//   grid: (timeTiles=690, chanBlocks=4, batch*bank=8), 256 threads = 8 waves.
//   Each wave owns a 16(ch) x 16(time) tile; 64 x wmma_f32_16x16x32_f16.
//   Filter chunks (16 x 256 f16, row stride 2048) are staged into LDS by the
//   Tensor Data Mover; the signal window is staged with b128 loads.
// Fused epilogue: bank 0 -> *g_lin ; bank 1 -> broken-stick compression.
// ===========================================================================
__global__ __launch_bounds__(256) void conv_wmma(
        const _Float16* __restrict__ xh,
        const _Float16* __restrict__ h16,
        const float*    __restrict__ params,
        float* __restrict__ ulin,
        float* __restrict__ unl) {
    __shared__ alignas(16) _Float16 xs[GTL + TTILE + 32];   // signal window
    __shared__ alignas(16) _Float16 hs[16 * 256];           // filter chunk

    const int tile = blockIdx.x;
    const int cb   = blockIdx.y;
    const int b    = blockIdx.z & 3;
    const int bank = blockIdx.z >> 2;
    const int tid  = threadIdx.x;
    const int wave = tid >> 5;
    const int lane = tid & 31;
    const int t0   = tile * TTILE;

    // stage signal window x[t0-2048 .. t0+127] (padded coords [t0, t0+2175])
    {
        const _Float16* src = xh + (size_t)b * TP + t0;
        for (int i = tid; i < (GTL + TTILE) / 8; i += 256)
            *(v8h*)&xs[i * 8] = *(const v8h*)&src[i * 8];
    }

    const int col = (wave << 4) + (lane & 15);   // time column within tile
    const int row = lane & 15;                   // A-matrix row (channel)
    const int hi  = lane >> 4;                   // lane half
    const int ko  = hi << 3;                     // A k-offset: 0 / 8
    const int kbB = hi << 3;                     // B k-offset: 0 / 8

    v8f acc = {};
    const _Float16* hbase =
        h16 + (size_t)(bank * NCHP + cb * 16) * GTL;

    for (int chunk = 0; chunk < 8; ++chunk) {
        __syncthreads();    // previous chunk fully consumed
        const _Float16* hsrc = hbase + chunk * 256;
#if HAVE_TDM
        if (wave == 0) {
            // TDM: 256x16 element tile, 2B elements, row stride 2048 elems
            tdm_load_2d_f16(lds_addr_of(hs), hsrc,
                            /*tile*/ 256, 16,
                            /*tensor*/ GTL, NCHP,
                            /*stride*/ GTL);
            __builtin_amdgcn_s_wait_tensorcnt(0);
        }
#else
        for (int i = tid; i < 512; i += 256) {
            int r = i >> 5;
            int cidx = (i & 31) * 8;
            *(v8h*)&hs[r * 256 + cidx] =
                *(const v8h*)&hsrc[(size_t)r * GTL + cidx];
        }
#endif
        if (chunk < 7 && tid < 16)
            __builtin_prefetch(hsrc + (size_t)tid * GTL + 256, 0, 1);
        __syncthreads();    // chunk visible to all waves

        #pragma unroll
        for (int kb = 0; kb < 8; ++kb) {
            const int kk = kb * 32;
            // A fragment: 16x32 f16 per documented VGPR layout
            v16h a;
            {
                v8h lo = *(const v8h*)&hs[row * 256 + kk + ko];
                v8h hv = *(const v8h*)&hs[row * 256 + kk + ko + 16];
                #pragma unroll
                for (int i = 0; i < 8; ++i) { a[i] = lo[i]; a[8 + i] = hv[i]; }
            }
            // B fragment: Toeplitz window B[k][n] = x[t0 + n - k]
            v16h bf;
            const int base = GTL + col - (chunk * 256 + kk + kbB);
            #pragma unroll
            for (int i = 0; i < 16; ++i) bf[i] = xs[base - i - ((i >> 3) << 3)];

            acc = __builtin_amdgcn_wmma_f32_16x16x32_f16(
                false, a, false, bf, (short)0, acc, false, false);
        }
    }

    // epilogue + store (D layout: VGPR r -> M=r+8*hi, N=lane&15)
    const int t = t0 + col;
    #pragma unroll
    for (int r = 0; r < 8; ++r) {
        int c = cb * 16 + r + (hi << 3);
        if (c < N_CH && t < TLEN) {
            float u = acc[r];
            size_t o = ((size_t)b * N_CH + c) * TLEN + t;
            if (bank == 0) {
                ulin[o] = u * params[c * 8 + 1];
            } else {
                float au = fabsf(u);
                float v = fminf(params[c * 8 + 2] * au,
                                params[c * 8 + 3] * powf(au + 1e-12f, 0.25f));
                unl[o] = copysignf(v, u);
            }
        }
    }
}

// ===========================================================================
// Kernel 5: per-(b,c) recurrences.
//   nl-path RBJ LP biquad -> sum -> IHC (relu^0.23, 1500 Hz 1st-order LP)
//   -> 5 adaptation loops. Writes yad in-place over ulin.
// ===========================================================================
__global__ void chain(const float* __restrict__ params,
                      float* __restrict__ ulin,
                      const float* __restrict__ unl) {
    int idx = blockIdx.x * blockDim.x + threadIdx.x;
    if (idx >= 4 * N_CH) return;
    int c = idx % N_CH;

    // RBJ lowpass at fc_nl, Q = 1/sqrt(2)
    float w0 = TWO_PI_F * params[c * 8 + 4] / FS_F;
    float cw = cosf(w0), sw = sinf(w0);
    float al = sw / (2.0f * 0.70710678f);
    float ia0 = 1.0f / (1.0f + al);
    float bb0 = (1.0f - cw) * 0.5f * ia0;
    float bb1 = (1.0f - cw) * ia0;
    float bb2 = bb0;
    float aa1 = -2.0f * cw * ia0;
    float aa2 = (1.0f - al) * ia0;
    float z1 = 0.0f, z2 = 0.0f;

    // IHC first-order LP @1500 Hz
    float aI = expf(-TWO_PI_F * 1500.0f / FS_F);
    float bI = 1.0f - aI;
    float vi = 0.0f;

    // adaptation loop constants
    const float taus[5] = {0.005f, 0.050f, 0.129f, 0.253f, 0.500f};
    float a1c[5], b0c[5], st[5], fac[5], efc[5], off[5];
    #pragma unroll
    for (int k = 0; k < 5; ++k) {
        a1c[k] = expf(-1.0f / (taus[k] * FS_F));
        b0c[k] = 1.0f - a1c[k];
        float s0 = powf(1e-5f, exp2f(-(float)(k + 1)));
        st[k] = s0;
        float maxv = (1.0f - s0 * s0) * 10.0f - 1.0f;
        fac[k] = 2.0f * maxv;
        efc[k] = -2.0f / maxv;
        off[k] = maxv - 1.0f;
    }
    float corr = powf(1e-5f, 0.03125f);
    float osc = 100.0f / (1.0f - corr);

    size_t base = (size_t)idx * TLEN;
    for (int t = 0; t < TLEN; ++t) {
        float v = unl[base + t];
        float y = bb0 * v + z1;
        z1 = bb1 * v + z2 - aa1 * y;
        z2 = bb2 * v - aa2 * y;
        float s = ulin[base + t] + y;
        float v2 = powf(fmaxf(s, 0.0f) + 1e-9f, 0.23f);
        vi = bI * v2 + aI * vi;
        float vv = fmaxf(vi, 1e-5f);
        #pragma unroll
        for (int k = 0; k < 5; ++k) {
            float tmp = vv / st[k];
            if (tmp > 1.0f)
                tmp = fac[k] / (1.0f + expf(efc[k] * (tmp - 1.0f))) - off[k];
            st[k] = a1c[k] * st[k] + b0c[k] * tmp;
            vv = tmp;
        }
        ulin[base + t] = (vv - corr) * osc;
    }
}

// ===========================================================================
// Kernel 6: windowed-sinc anti-alias + decimate by 4.
// ===========================================================================
__global__ __launch_bounds__(256) void resample4(const float* __restrict__ yad,
                                                 float* __restrict__ y2) {
    __shared__ float h[65];
    __shared__ float hsum;
    int tid = threadIdx.x;
    if (tid < 65) {
        float nn = (float)(tid - 32);
        float sx = (tid == 32) ? 1.0f
                               : sinf(PI_F * nn * 0.25f) / (PI_F * nn * 0.25f);
        float w = 0.5f - 0.5f * cosf(TWO_PI_F * (float)tid / 64.0f);
        h[tid] = 0.25f * sx * w;
    }
    __syncthreads();
    if (tid == 0) {
        float s = 0.0f;
        for (int i = 0; i < 65; ++i) s += h[i];
        hsum = s;
    }
    __syncthreads();
    float inv = 1.0f / hsum;

    size_t idx = (size_t)blockIdx.x * 256 + tid;
    if (idx >= (size_t)4 * N_CH * T2LEN) return;
    size_t rowi = idx / T2LEN;
    int t2 = (int)(idx - rowi * T2LEN);
    const float* src = yad + rowi * TLEN;
    int tb = 4 * t2 - 32;
    float s = 0.0f;
    for (int k = 0; k < 65; ++k) {
        int ti = tb + k;
        if (ti >= 0 && ti < TLEN) s += h[k] * src[ti];
    }
    y2[idx] = s * inv;
}

// ===========================================================================
// Kernel 7: modulation filterbank (8 biquads) + fc/4 mask -> d_out [B,C,M,T2]
// ===========================================================================
__global__ void modfb(const float* __restrict__ params,
                      const float* __restrict__ y2,
                      float* __restrict__ out) {
    int idx = blockIdx.x * blockDim.x + threadIdx.x;
    if (idx >= 4 * N_CH * 8) return;
    int m = idx & 7;
    int bc = idx >> 3;
    int c = bc % N_CH;
    float fs2 = FS_F * 0.25f;

    float mfc = (m == 0) ? 0.0f : 5.0f * powf(5.0f / 3.0f, (float)(m - 1));
    float f0, Q;
    bool lp = (m == 0);
    if (lp) { f0 = 2.5f; Q = 0.70710678f; }
    else    { f0 = mfc; Q = (mfc <= 10.0f) ? fmaxf(mfc, 5.0f) / 5.0f : 2.0f; }

    float w0 = TWO_PI_F * f0 / fs2;
    float cw = cosf(w0), sw = sinf(w0);
    float al = sw / (2.0f * Q);
    float ia0 = 1.0f / (1.0f + al);
    float b0, b1, b2;
    if (lp) { b0 = (1.0f - cw) * 0.5f * ia0; b1 = (1.0f - cw) * ia0; b2 = b0; }
    else    { b0 = al * ia0; b1 = 0.0f; b2 = -al * ia0; }
    float a1 = -2.0f * cw * ia0;
    float a2 = (1.0f - al) * ia0;

    float mask = (mfc <= 0.25f * params[c * 8 + 0]) ? 1.0f : 0.0f;
    const float* src = y2 + (size_t)bc * T2LEN;
    float* dst = out + ((size_t)bc * 8 + m) * T2LEN;

    float z1 = 0.0f, z2 = 0.0f;
    for (int t = 0; t < T2LEN; ++t) {
        float xv = src[t];
        float y = b0 * xv + z1;
        z1 = b1 * xv + z2 - a1 * y;
        z2 = b2 * xv - a2 * y;
        dst[t] = y * mask;
    }
}

// ===========================================================================
// host launcher
// ===========================================================================
extern "C" void kernel_launch(void* const* d_in, const int* in_sizes, int n_in,
                              void* d_out, int out_size, void* d_ws, size_t ws_size,
                              hipStream_t stream) {
    (void)in_sizes; (void)n_in; (void)out_size; (void)ws_size;
    const float* x = (const float*)d_in[0];
    float* out = (float*)d_out;
    char* ws = (char*)d_ws;

    size_t off = 0;
    auto take = [&](size_t bytes) -> char* {
        char* p = ws + off;
        off = (off + bytes + 255) & ~(size_t)255;
        return p;
    };
    float*    params = (float*)   take((size_t)N_CH * 8 * sizeof(float));
    _Float16* h16    = (_Float16*)take((size_t)2 * NCHP * GTL * sizeof(_Float16));
    _Float16* xh     = (_Float16*)take((size_t)4 * TP * sizeof(_Float16));
    float*    ulin   = (float*)   take((size_t)4 * N_CH * TLEN * sizeof(float));
    float*    unl    = (float*)   take((size_t)4 * N_CH * TLEN * sizeof(float));
    float*    y2     = (float*)   take((size_t)4 * N_CH * T2LEN * sizeof(float));

    param_gen<<<1, 64, 0, stream>>>(params);
    fir_gen<<<dim3(NCHP, 2), 256, 0, stream>>>(h16);
    {
        int total = 4 * TP;
        xpad<<<(total + 255) / 256, 256, 0, stream>>>(x, xh);
    }
    conv_wmma<<<dim3((TLEN + TTILE - 1) / TTILE, 4, 8), 256, 0, stream>>>(
        xh, h16, params, ulin, unl);
    chain<<<(4 * N_CH + 63) / 64, 64, 0, stream>>>(params, ulin, unl);
    {
        size_t total = (size_t)4 * N_CH * T2LEN;
        resample4<<<(unsigned)((total + 255) / 256), 256, 0, stream>>>(ulin, y2);
    }
    modfb<<<(4 * N_CH * 8 + 255) / 256, 256, 0, stream>>>(params, y2, out);
}
